// TSM_multiKH_22084721836645
// MI455X (gfx1250) — matrix operand order repaired and checked by reading, not compiled, not run on hardware
//
#include <hip/hip_runtime.h>
#include <cmath>

// ---------------------------------------------------------------------------
// CDNA5 (gfx1250, wave32) implementation.
constexpr int CB   = 32;    // batch
constexpr int CROI = 200;   // ROI
constexpr int CT   = 512;   // T
constexpr int CK   = 32;    // NK*NH output channels
constexpr int CDP1 = 208;   // d=200 padded to 16 (k1 feature dim)
constexpr int CSP2 = 224;   // ROI padded to 32 (k2 sequence dim / gram tiles)
constexpr int CDFF = 128;
constexpr int CCAT = 416;   // 2*ROI padded to 32 (K dim of l0 GEMM)
constexpr int CTB  = CSP2 / 32;                 // 7  (32x32 gram blocks per dim)
constexpr int CUT  = CTB * (CTB + 1) / 2;       // 28 upper-triangular blocks

typedef __attribute__((ext_vector_type(16))) __bf16 v16bf;
typedef __attribute__((ext_vector_type(8)))  float  v8f;

// ---------------------------------------------------------------------------
// WMMA fragment loaders (bf16 16x16x32, wave32; this code's contiguous k order, same in A and B)
// A (16x32): lane<16 holds row m, K {0..15}; lane>=16 K {16..31}
// B (32x16): lane<16 holds col n, K {0..15}; lane>=16 K {16..31}
static __device__ __forceinline__ v16bf frag_a_load(const float* base, long stride,
                                                    int k0, int half, int kmax) {
  v16bf r;
  int kb = k0 + half * 16;
#pragma unroll
  for (int e = 0; e < 8; ++e) {
    int ka = kb + e;
    int kc = kb + 8 + e;
    float f0 = (ka < kmax) ? base[(long)ka * stride] : 0.0f;
    float f1 = (kc < kmax) ? base[(long)kc * stride] : 0.0f;
    r[e]     = (__bf16)f0;
    r[e + 8] = (__bf16)f1;
  }
  return r;
}

static __device__ __forceinline__ v16bf frag_b_load(const float* base, long stride,
                                                    int k0, int half, int kmax) {
  v16bf r;
  int kb = k0 + half * 16;
#pragma unroll
  for (int e = 0; e < 16; ++e) {
    int ka = kb + e;
    float f = (ka < kmax) ? base[(long)ka * stride] : 0.0f;
    r[e] = (__bf16)f;
  }
  return r;
}

// Vectorized fast paths (full 32-K chunk in-bounds, unit stride, 16B aligned).
static __device__ __forceinline__ v16bf frag_a_vec(const float* base, int k0, int half) {
  float av[16];
  int kb = k0 + half * 16;
  *(float4*)&av[0]  = *(const float4*)(base + kb);
  *(float4*)&av[4]  = *(const float4*)(base + kb + 4);
  *(float4*)&av[8]  = *(const float4*)(base + kb + 8);
  *(float4*)&av[12] = *(const float4*)(base + kb + 12);
  v16bf r;
#pragma unroll
  for (int e = 0; e < 8; ++e) { r[e] = (__bf16)av[e]; r[e + 8] = (__bf16)av[8 + e]; }
  return r;
}

static __device__ __forceinline__ v16bf frag_b_vec(const float* base, int k0, int half) {
  float bv[16];
  int kb = k0 + half * 16;
  *(float4*)&bv[0]  = *(const float4*)(base + kb);
  *(float4*)&bv[4]  = *(const float4*)(base + kb + 4);
  *(float4*)&bv[8]  = *(const float4*)(base + kb + 8);
  *(float4*)&bv[12] = *(const float4*)(base + kb + 12);
  v16bf r;
#pragma unroll
  for (int e = 0; e < 16; ++e) r[e] = (__bf16)bv[e];
  return r;
}

// ---------------------------------------------------------------------------
// Generic batched GEMM:  C[m,n] = act(A[m,:] . W[n,:] + bias[n]) + R[m,n]
//   BKN==0: W row-major (W[n,k] at n*ldw + k)   BKN==1: W K-major (k*ldw + n)
//   ACT: 0 none, 1 relu, 2 relu(tanh)
// Block = 4 waves covering 32 (M) x 64 (N); each wave does 16x32 (two N tiles)
// so each A fragment load+convert feeds two WMMAs.
template <int ACT, int BKN>
__global__ __launch_bounds__(128) void gemm_bf16(
    const float* __restrict__ A, const float* __restrict__ W,
    const float* __restrict__ bias, const float* __restrict__ R,
    float* __restrict__ C, int Mwrite, int Nwrite, int Nreal, int Kreal,
    int lda, int ldw, int ldc,
    long aZ, long aH, long wZ, long wH, long cZ, long cH, int nh) {
  int z = blockIdx.z;
  int bz = z / nh, hz = z - bz * nh;
  const float* Ab = A + (long)bz * aZ + (long)hz * aH;
  const float* Wb = W + (long)bz * wZ + (long)hz * wH;
  long coff = (long)bz * cZ + (long)hz * cH;
  float* Cb = C + coff;

  int tid = threadIdx.x;
  int wave = tid >> 5, lane = tid & 31;
  int lm = lane & 15, half = lane >> 4;
  int row0 = blockIdx.y * 32 + (wave >> 1) * 16;
  int colB = blockIdx.x * 64 + (wave & 1) * 32;
  int m = row0 + lm;

  const float* arow = Ab + (long)m * lda;
  int akmax = (m < Mwrite) ? Kreal : 0;
  long wstride = BKN ? (long)ldw : 1L;

  const float* wbase[2];
  int wkmax[2], ncol[2];
#pragma unroll
  for (int jt = 0; jt < 2; ++jt) {
    int n = colB + jt * 16 + lm;
    ncol[jt] = n;
    bool vv = (n < Nreal);
    wbase[jt] = vv ? (BKN ? (Wb + n) : (Wb + (long)n * ldw)) : Wb;
    wkmax[jt] = vv ? Kreal : 0;
  }

  v8f zero = {};
  v8f acc[2] = {zero, zero};
  int kceil = (Kreal + 31) & ~31;
  for (int k0 = 0; k0 < kceil; k0 += 32) {
    v16bf af;
    if (k0 + 32 <= akmax) af = frag_a_vec(arow, k0, half);
    else                  af = frag_a_load(arow, 1L, k0, half, akmax);
#pragma unroll
    for (int jt = 0; jt < 2; ++jt) {
      v16bf wf;
      if (BKN == 0 && k0 + 32 <= wkmax[jt]) wf = frag_b_vec(wbase[jt], k0, half);
      else                                  wf = frag_b_load(wbase[jt], wstride, k0, half, wkmax[jt]);
      acc[jt] = __builtin_amdgcn_wmma_f32_16x16x32_bf16(false, af, false, wf, (short)0,
                                                        acc[jt], false, false);
    }
  }

#pragma unroll
  for (int jt = 0; jt < 2; ++jt) {
    int n = ncol[jt];
    if (n < Nwrite) {
      float bval = (bias != nullptr && n < Nreal) ? bias[n] : 0.0f;
#pragma unroll
      for (int v = 0; v < 8; ++v) {
        int r = row0 + v + half * 8;  // C layout: lanes0-15 rows v, lanes16-31 rows v+8
        if (r < Mwrite) {
          float val = acc[jt][v] + bval;
          if (ACT == 1) val = fmaxf(val, 0.0f);
          if (ACT == 2) val = fmaxf(tanhf(val), 0.0f);
          if (R != nullptr) val += R[coff + (long)r * ldc + n];
          Cb[(long)r * ldc + n] = val;
        }
      }
    }
  }
}

// ---------------------------------------------------------------------------
// LayerNorm over last dim d, padded width dp; rows beyond Sreal are zeroed.
__global__ __launch_bounds__(256) void ln_kernel(const float* __restrict__ in,
                                                 float* __restrict__ out,
                                                 const float* __restrict__ g,
                                                 const float* __restrict__ bvec,
                                                 int d, int dp, int Sp, int Sreal) {
  __shared__ float s1[256], s2[256];
  int row = blockIdx.x, tid = threadIdx.x;
  int srow = row % Sp;
  const float* xi = in + (long)row * dp;
  float* yo = out + (long)row * dp;
  if (srow >= Sreal) {
    for (int c = tid; c < dp; c += 256) yo[c] = 0.0f;
    return;
  }
  float ls = 0.0f, lq = 0.0f;
  for (int c = tid; c < d; c += 256) {
    float v = xi[c];
    ls += v; lq += v * v;
  }
  s1[tid] = ls; s2[tid] = lq;
  __syncthreads();
  for (int st = 128; st > 0; st >>= 1) {
    if (tid < st) { s1[tid] += s1[tid + st]; s2[tid] += s2[tid + st]; }
    __syncthreads();
  }
  float mean = s1[0] / d;
  float var = s2[0] / d - mean * mean;
  float rstd = rsqrtf(var + 1e-5f);
  for (int c = tid; c < dp; c += 256)
    yo[c] = (c < d) ? ((xi[c] - mean) * rstd * g[c] + bvec[c]) : 0.0f;
}

// ---------------------------------------------------------------------------
// Row softmax over padded SpxSp score matrices; masks padded keys/rows to 0.
__global__ __launch_bounds__(256) void softmax_kernel(float* __restrict__ att,
                                                      int Sp, int Sreal, float scale) {
  __shared__ float red[256];
  long base = (long)blockIdx.x * Sp;
  int srow = blockIdx.x % Sp;
  int tid = threadIdx.x;
  float* rowp = att + base;
  if (srow >= Sreal) {
    for (int j = tid; j < Sp; j += 256) rowp[j] = 0.0f;
    return;
  }
  float lmax = -3.0e38f;
  for (int j = tid; j < Sreal; j += 256) lmax = fmaxf(lmax, rowp[j] * scale);
  red[tid] = lmax; __syncthreads();
  for (int st = 128; st > 0; st >>= 1) {
    if (tid < st) red[tid] = fmaxf(red[tid], red[tid + st]);
    __syncthreads();
  }
  float m = red[0];
  __syncthreads();
  float lsum = 0.0f;
  for (int j = tid; j < Sreal; j += 256) lsum += __expf(rowp[j] * scale - m);
  red[tid] = lsum; __syncthreads();
  for (int st = 128; st > 0; st >>= 1) {
    if (tid < st) red[tid] += red[tid + st];
    __syncthreads();
  }
  float inv = 1.0f / red[0];
  for (int j = tid; j < Sp; j += 256)
    rowp[j] = (j < Sreal) ? __expf(rowp[j] * scale - m) * inv : 0.0f;
}

// ---------------------------------------------------------------------------
// s[b,k] = sum_t tpk[b,t,k]
__global__ __launch_bounds__(128) void colsum_kernel(const float* __restrict__ tpk,
                                                     float* __restrict__ sb) {
  __shared__ float red[128];
  int bk = blockIdx.x;
  int b = bk >> 5, k = bk & 31;
  int tid = threadIdx.x;
  float ls = 0.0f;
  for (int t = tid; t < CT; t += 128) ls += tpk[((long)b * CT + t) * CK + k];
  red[tid] = ls; __syncthreads();
  for (int st = 64; st > 0; st >>= 1) {
    if (tid < st) red[tid] += red[tid + st];
    __syncthreads();
  }
  if (tid == 0) sb[bk] = red[0];
}

// ---------------------------------------------------------------------------
// Layout helpers
__global__ __launch_bounds__(256) void xT_kernel(const float* __restrict__ x,
                                                 float* __restrict__ xT) {
  long idx = (long)blockIdx.x * 256 + threadIdx.x;
  if (idx >= (long)CB * CT * CDP1) return;
  int i = (int)(idx % CDP1);
  long r = idx / CDP1;
  int t = (int)(r % CT);
  int b = (int)(r / CT);
  xT[idx] = (i < CROI) ? x[((long)b * CROI + i) * CT + t] : 0.0f;
}

__global__ __launch_bounds__(256) void xpad_kernel(const float* __restrict__ x,
                                                   float* __restrict__ xp) {
  long idx = (long)blockIdx.x * 256 + threadIdx.x;
  if (idx >= (long)CB * CSP2 * CT) return;
  int t = (int)(idx % CT);
  long r = idx / CT;
  int i = (int)(r % CSP2);
  int b = (int)(r / CSP2);
  xp[idx] = (i < CROI) ? x[((long)b * CROI + i) * CT + t] : 0.0f;
}

// One-time bf16 hi/lo split of padded x (B side of the gram is unweighted).
__global__ __launch_bounds__(256) void split_kernel(const float* __restrict__ xp,
                                                    __bf16* __restrict__ xh,
                                                    __bf16* __restrict__ xl) {
  long idx = (long)blockIdx.x * 256 + threadIdx.x;
  if (idx >= (long)CB * CSP2 * CT) return;
  float v = xp[idx];
  __bf16 h = (__bf16)v;
  xh[idx] = h;
  xl[idx] = (__bf16)(v - (float)h);
}

__global__ __launch_bounds__(256) void concat_kernel(const float* __restrict__ e1,
                                                     const float* __restrict__ e2,
                                                     float* __restrict__ tpc) {
  long idx = (long)blockIdx.x * 256 + threadIdx.x;
  if (idx >= (long)CB * CT * CCAT) return;
  int c = (int)(idx % CCAT);
  long r = idx / CCAT;
  int t = (int)(r % CT);
  int b = (int)(r / CT);
  float v = 0.0f;
  if (c < CROI) v = e1[((long)b * CT + t) * CDP1 + c];
  else if (c < 2 * CROI) v = e2[((long)b * CSP2 + (c - CROI)) * CT + t];
  tpc[idx] = v;
}

// ---------------------------------------------------------------------------
// Weighted Gram: c[b,k,i,j] = (1/s[b,k]) * sum_t x[b,i,t]*w[b,t,k]*x[b,j,t]
// Symmetric in (i,j): only upper-triangular 32x32 blocks computed (28 of 49),
// each result stored to [i,j] and [j,i].
// bf16 hi/lo split (AhBh + AhBl + AlBh) for near-fp32 accuracy.
// Block = 2 waves; each wave computes a 16x32 strip: one weighted A conversion
// feeds SIX WMMAs (two B column tiles x 3 hi/lo terms) on independent chains.
__global__ __launch_bounds__(64) void gram_kernel(const float* __restrict__ xp,
                                                  const __bf16* __restrict__ xh,
                                                  const __bf16* __restrict__ xl,
                                                  const float* __restrict__ tpk,
                                                  const float* __restrict__ sb,
                                                  float* __restrict__ out) {
  __shared__ float wsh[CT];
  int b = blockIdx.z, k = blockIdx.y;
  int tid = threadIdx.x;
  for (int q = tid; q < CT; q += 64)
    wsh[q] = tpk[((long)b * CT + q) * CK + k];
  __syncthreads();

  // Map linear block id -> upper-triangular (bi, bj), bi <= bj.
  int rem = blockIdx.x, bi = 0;
  while (rem >= CTB - bi) { rem -= CTB - bi; ++bi; }
  int bj = bi + rem;

  int wave = tid >> 5, lane = tid & 31;
  int lm = lane & 15, half = lane >> 4;
  int i0 = bi * 32 + wave * 16;
  int j0 = bj * 32;

  const float*  arow = xp + ((long)b * CSP2 + i0 + lm) * CT;
  const __bf16* b0h  = xh + ((long)b * CSP2 + j0 + lm) * CT;
  const __bf16* b0l  = xl + ((long)b * CSP2 + j0 + lm) * CT;
  const __bf16* b1h  = xh + ((long)b * CSP2 + j0 + 16 + lm) * CT;
  const __bf16* b1l  = xl + ((long)b * CSP2 + j0 + 16 + lm) * CT;

  v8f z = {};
  v8f a00 = z, a01 = z, a02 = z, a10 = z, a11 = z, a12 = z;
  for (int t0 = 0; t0 < CT; t0 += 32) {
    int ta = t0 + half * 16;
    float av[16];
    *(float4*)&av[0]  = *(const float4*)(arow + ta);
    *(float4*)&av[4]  = *(const float4*)(arow + ta + 4);
    *(float4*)&av[8]  = *(const float4*)(arow + ta + 8);
    *(float4*)&av[12] = *(const float4*)(arow + ta + 12);
    int tb = t0 + half * 16;
    v16bf f0h = *(const v16bf*)(b0h + tb);
    v16bf f0l = *(const v16bf*)(b0l + tb);
    v16bf f1h = *(const v16bf*)(b1h + tb);
    v16bf f1l = *(const v16bf*)(b1l + tb);
    if (t0 + 32 < CT) {
      __builtin_prefetch(arow + ta + 32, 0, 1);
      __builtin_prefetch(b0h + tb + 32, 0, 1);
      __builtin_prefetch(b1h + tb + 32, 0, 1);
    }
    v16bf ah, al;
#pragma unroll
    for (int e = 0; e < 8; ++e) {
      float wa0 = av[e] * wsh[ta + e];
      float wa1 = av[8 + e] * wsh[ta + 8 + e];
      __bf16 h0 = (__bf16)wa0, h1 = (__bf16)wa1;
      ah[e]     = h0; al[e]     = (__bf16)(wa0 - (float)h0);
      ah[e + 8] = h1; al[e + 8] = (__bf16)(wa1 - (float)h1);
    }
    a00 = __builtin_amdgcn_wmma_f32_16x16x32_bf16(false, ah, false, f0h, (short)0, a00, false, false);
    a10 = __builtin_amdgcn_wmma_f32_16x16x32_bf16(false, ah, false, f1h, (short)0, a10, false, false);
    a01 = __builtin_amdgcn_wmma_f32_16x16x32_bf16(false, ah, false, f0l, (short)0, a01, false, false);
    a11 = __builtin_amdgcn_wmma_f32_16x16x32_bf16(false, ah, false, f1l, (short)0, a11, false, false);
    a02 = __builtin_amdgcn_wmma_f32_16x16x32_bf16(false, al, false, f0h, (short)0, a02, false, false);
    a12 = __builtin_amdgcn_wmma_f32_16x16x32_bf16(false, al, false, f1h, (short)0, a12, false, false);
  }

  float inv = 1.0f / sb[b * CK + k];
  long obase = ((long)b * CK + k) * CROI;
  int j1 = j0 + lm;
  int j2 = j0 + 16 + lm;
#pragma unroll
  for (int v = 0; v < 8; ++v) {
    int i = i0 + v + half * 8;
    if (i < CROI) {
      if (j1 < CROI) {
        float val = (a00[v] + a01[v] + a02[v]) * inv;
        out[(obase + i) * CROI + j1] = val;
        out[(obase + j1) * CROI + i] = val;   // symmetric mirror
      }
      if (j2 < CROI) {
        float val = (a10[v] + a11[v] + a12[v]) * inv;
        out[(obase + i) * CROI + j2] = val;
        out[(obase + j2) * CROI + i] = val;   // symmetric mirror
      }
    }
  }
}

// ---------------------------------------------------------------------------
struct EncP {
  const float *in_w, *in_b, *out_w, *out_b, *ln1_g, *ln1_b, *ln2_g, *ln2_b,
              *ff1_w, *ff1_b, *ff2_w, *ff2_b;
};

static void run_encoder(const float* inBuf, float* outBuf, float* hb, float* qkv,
                        float* att, float* ob, float* xr, float* ffb, const EncP& p,
                        int Sp, int Sreal, int d, int dp, int dh, int d3, int d3p,
                        hipStream_t stream) {
  int Mtot = CB * Sp;   // multiple of 32 by construction
  float scale = 1.0f / sqrtf((float)dh);

  // h = LN1(x)
  ln_kernel<<<Mtot, 256, 0, stream>>>(inBuf, hb, p.ln1_g, p.ln1_b, d, dp, Sp, Sreal);
  // qkv = h @ in_w^T + in_b
  {
    dim3 g((d3p + 63) / 64, Mtot / 32, 1);
    gemm_bf16<0, 0><<<g, 128, 0, stream>>>(hb, p.in_w, p.in_b, nullptr, qkv,
        Mtot, d3p, d3, d, dp, d, d3p, 0, 0, 0, 0, 0, 0, 1);
  }
  // scores[b,h] = q @ k^T
  {
    dim3 g((Sp + 63) / 64, (Sp + 31) / 32, CB * 2);
    gemm_bf16<0, 0><<<g, 128, 0, stream>>>(qkv, qkv + d, nullptr, nullptr, att,
        Sp, Sp, Sp, dh, d3p, d3p, Sp,
        (long)Sp * d3p, (long)dh, (long)Sp * d3p, (long)dh,
        2L * Sp * Sp, (long)Sp * Sp, 2);
  }
  softmax_kernel<<<CB * 2 * Sp, 256, 0, stream>>>(att, Sp, Sreal, scale);
  // o[b,h] = att @ v   (v accessed K-major)
  {
    dim3 g((dh + 63) / 64, (Sp + 31) / 32, CB * 2);
    gemm_bf16<0, 1><<<g, 128, 0, stream>>>(att, qkv + 2 * d, nullptr, nullptr, ob,
        Sp, dh, dh, Sp, Sp, d3p, dp,
        2L * Sp * Sp, (long)Sp * Sp, (long)Sp * d3p, (long)dh,
        (long)Sp * dp, (long)dh, 2);
  }
  // x = x + o @ out_w^T + out_b
  {
    dim3 g((dp + 63) / 64, Mtot / 32, 1);
    gemm_bf16<0, 0><<<g, 128, 0, stream>>>(ob, p.out_w, p.out_b, inBuf, xr,
        Mtot, dp, d, d, dp, d, dp, 0, 0, 0, 0, 0, 0, 1);
  }
  // h2 = LN2(x)  (reuse ob)
  ln_kernel<<<Mtot, 256, 0, stream>>>(xr, ob, p.ln2_g, p.ln2_b, d, dp, Sp, Sreal);
  // ff = relu(h2 @ ff1^T + b1)
  {
    dim3 g((CDFF + 63) / 64, Mtot / 32, 1);
    gemm_bf16<1, 0><<<g, 128, 0, stream>>>(ob, p.ff1_w, p.ff1_b, nullptr, ffb,
        Mtot, CDFF, CDFF, d, dp, d, CDFF, 0, 0, 0, 0, 0, 0, 1);
  }
  // out = x + ff @ ff2^T + b2
  {
    dim3 g((dp + 63) / 64, Mtot / 32, 1);
    gemm_bf16<0, 0><<<g, 128, 0, stream>>>(ffb, p.ff2_w, p.ff2_b, xr, outBuf,
        Mtot, dp, d, CDFF, CDFF, CDFF, dp, 0, 0, 0, 0, 0, 0, 1);
  }
}

extern "C" void kernel_launch(void* const* d_in, const int* in_sizes, int n_in,
                              void* d_out, int out_size, void* d_ws, size_t ws_size,
                              hipStream_t stream) {
  (void)in_sizes; (void)n_in; (void)out_size; (void)ws_size;
  const float* x = (const float*)d_in[0];
  auto f = [&](int i) { return (const float*)d_in[i]; };
  EncP p1{f(1), f(2), f(3), f(4), f(5), f(6), f(7), f(8), f(9), f(10), f(11), f(12)};
  EncP p2{f(13), f(14), f(15), f(16), f(17), f(18), f(19), f(20), f(21), f(22), f(23), f(24)};
  const float* l0w = f(25);
  const float* l0b = f(26);
  float* out = (float*)d_out;

  // Workspace layout (floats). Shared arenas sized for the larger encoder.
  float* ws = (float*)d_ws;
  size_t off = 0;
  auto alloc = [&](size_t n) { float* p = ws + off; off += n; return p; };
  float* xT   = alloc((size_t)CB * CT * CDP1);     // (B,512,208) x transposed+padded
  float* xp   = alloc((size_t)CB * CSP2 * CT);     // (B,224,512) x row-padded
  float* hb   = alloc((size_t)CB * CSP2 * CT);     // LN output arena
  float* qkv  = alloc((size_t)CB * CSP2 * 1536);   // >= (B,512,608)
  float* att  = alloc((size_t)CB * 2 * CT * CT);   // >= (B,2,224,224)
  float* ob   = alloc((size_t)CB * CSP2 * CT);     // attn-out / h2 arena
  float* xr   = alloc((size_t)CB * CSP2 * CT);     // residual arena
  float* ffb  = alloc((size_t)CB * CT * CDFF);
  float* e1   = alloc((size_t)CB * CT * CDP1);     // encoder1 out (B,512,208)
  float* e2   = alloc((size_t)CB * CSP2 * CT);     // encoder2 out (B,224,512)
  float* tpc  = alloc((size_t)CB * CT * CCAT);     // concat (B,512,416)
  float* tpkb = alloc((size_t)CB * CT * CK);       // (B,512,32)
  float* sb   = alloc((size_t)CB * CK);
  __bf16* xh = (__bf16*)alloc((size_t)CB * CSP2 * CT / 2);  // bf16 hi arena
  __bf16* xl = (__bf16*)alloc((size_t)CB * CSP2 * CT / 2);  // bf16 lo arena

  {
    long n1 = (long)CB * CT * CDP1;
    xT_kernel<<<dim3((unsigned)((n1 + 255) / 256)), 256, 0, stream>>>(x, xT);
    long n2 = (long)CB * CSP2 * CT;
    xpad_kernel<<<dim3((unsigned)((n2 + 255) / 256)), 256, 0, stream>>>(x, xp);
    split_kernel<<<dim3((unsigned)((n2 + 255) / 256)), 256, 0, stream>>>(xp, xh, xl);
  }

  // Encoder 1: seq=512, d=200 (dp=208, dh=100, 3d=600->608)
  run_encoder(xT, e1, hb, qkv, att, ob, xr, ffb, p1, CT, CT, 200, CDP1, 100, 600, 608, stream);
  // Encoder 2: seq=200->224, d=512 (dh=256, 3d=1536)
  run_encoder(xp, e2, hb, qkv, att, ob, xr, ffb, p2, CSP2, CROI, CT, CT, 256, 1536, 1536, stream);

  {
    long n3 = (long)CB * CT * CCAT;
    concat_kernel<<<dim3((unsigned)((n3 + 255) / 256)), 256, 0, stream>>>(e1, e2, tpc);
  }
  // tp = relu(tanh(tpcat @ l0_w^T + l0_b)) : M=16384, N=32, K=400
  {
    dim3 g(1, (CB * CT) / 32, 1);
    gemm_bf16<2, 0><<<g, 128, 0, stream>>>(tpc, l0w, l0b, nullptr, tpkb,
        CB * CT, CK, CK, 2 * CROI, CCAT, 2 * CROI, CK, 0, 0, 0, 0, 0, 0, 1);
  }
  colsum_kernel<<<CB * CK, 128, 0, stream>>>(tpkb, sb);
  // c[b,k] = (X diag(w) X^T) / s : 28 upper-tri 32x32 blocks x K x B
  gram_kernel<<<dim3(CUT, CK, CB), 64, 0, stream>>>(xp, xh, xl, tpkb, sb, out);
}
